// TK_v6_42674795053885
// MI455X (gfx1250) — matrix operand order repaired and checked by reading, not compiled, not run on hardware
//
#include <hip/hip_runtime.h>
#include <hip/hip_bf16.h>
#include <math.h>

typedef __attribute__((ext_vector_type(16))) _Float16 v16h;
typedef __attribute__((ext_vector_type(8)))  _Float16 v8h;
typedef __attribute__((ext_vector_type(8)))  float    v8f;
typedef __attribute__((ext_vector_type(4)))  int      v4i;

// CDNA5 async global->LDS path (guarded; falls back to sync staging if absent)
#if defined(__AMDGCN__) && __has_builtin(__builtin_amdgcn_global_load_async_to_lds_b128) && __has_builtin(__builtin_amdgcn_s_wait_asynccnt)
#define TK_ASYNC 1
#else
#define TK_ASYNC 0
#endif

// RBF kernel constants: MU, 1/(2*sigma^2)
__constant__ float c_MU[11]  = {1.f,0.9f,0.7f,0.5f,0.3f,0.1f,-0.1f,-0.3f,-0.5f,-0.7f,-0.9f};
__constant__ float c_I2S[11] = {500000.f,50.f,50.f,50.f,50.f,50.f,50.f,50.f,50.f,50.f,50.f};

__device__ __forceinline__ v8f zero8(){ v8f z;
#pragma unroll
  for (int i=0;i<8;i++) z[i]=0.f; return z; }

__device__ __forceinline__ v8f wmma16(v16h a, v16h b, v8f c){
  return __builtin_amdgcn_wmma_f32_16x16x32_f16(false, a, false, b, (short)0, c, false, false);
}

// A fragment (16x32, MxK): lane m=l&15; elems 0..15 -> K=kh..kh+15, kh=16*(l>>4)
__device__ __forceinline__ v16h load_a(const _Float16* __restrict__ p, int ld, int lane){
  int m  = lane & 15;
  int kh = (lane >> 4) << 4;
  const _Float16* q = p + (size_t)m * ld + kh;
  v16h f;
#pragma unroll
  for (int i = 0; i < 16; ++i) f[i] = q[i];
  return f;
}

// B fragment (32x16, KxN) from N-major (row=N, K contiguous) array: lane n=l&15; K=16*(l>>4)+i
__device__ __forceinline__ v16h load_b_nk(const _Float16* __restrict__ p, int ld, int lane){
  int n  = lane & 15;
  int kb = (lane >> 4) << 4;
  const _Float16* q = p + (size_t)n * ld + kb;
  v16h f;
#pragma unroll
  for (int i = 0; i < 16; ++i) f[i] = q[i];
  return f;
}

// B fragment from K-major (row=K, N contiguous) array, e.g. V[K][N]: strided reads
__device__ __forceinline__ v16h load_b_kn(const _Float16* __restrict__ p, int ld, int lane){
  int n  = lane & 15;
  int kb = (lane >> 4) << 4;
  const _Float16* q = p + (size_t)kb * ld + n;
  v16h f;
#pragma unroll
  for (int i = 0; i < 16; ++i) f[i] = q[(size_t)i * ld];
  return f;
}

// one b128 global->LDS stage (async if available)
__device__ __forceinline__ void stage_b128(_Float16* dst, const _Float16* src){
#if TK_ASYNC
  __builtin_amdgcn_global_load_async_to_lds_b128(
      (__attribute__((address_space(1))) v4i*)src,
      (__attribute__((address_space(3))) v4i*)dst, 0, 0);
#else
  *(v8h*)dst = *(const v8h*)src;
#endif
}
__device__ __forceinline__ void stage_drain(){
#if TK_ASYNC
  __builtin_amdgcn_s_wait_asynccnt(0);
#endif
}

// ---------------- fp32 -> f16 convert ----------------
__global__ void k_f32_to_f16(const float* __restrict__ a, _Float16* __restrict__ o, long long n){
  long long i = (long long)blockIdx.x * blockDim.x + threadIdx.x;
  long long st = (long long)gridDim.x * blockDim.x;
  for (; i < n; i += st) o[i] = (_Float16)a[i];
}

// ---------------- add-pos (optional) + LayerNorm; writes fp32 + f16 ----------------
__global__ void k_addpos_ln(const float* __restrict__ in, const float* __restrict__ pos, int S,
                            const float* __restrict__ g, const float* __restrict__ bb,
                            float* __restrict__ out32, _Float16* __restrict__ out16){
  __shared__ float red[256];
  int t = blockIdx.x, tid = threadIdx.x;
  const float* row = in + (size_t)t * 512;
  float x0 = row[tid], x1 = row[tid + 256];
  if (pos){ int s = t % S; x0 += pos[(size_t)s*512 + tid]; x1 += pos[(size_t)s*512 + tid + 256]; }
  red[tid] = x0 + x1; __syncthreads();
  for (int o = 128; o > 0; o >>= 1){ if (tid < o) red[tid] += red[tid + o]; __syncthreads(); }
  float mean = red[0] * (1.f/512.f);
  __syncthreads();
  float d0 = x0 - mean, d1 = x1 - mean;
  red[tid] = d0*d0 + d1*d1; __syncthreads();
  for (int o = 128; o > 0; o >>= 1){ if (tid < o) red[tid] += red[tid + o]; __syncthreads(); }
  float inv = rsqrtf(red[0] * (1.f/512.f) + 1e-5f);
  float y0 = d0 * inv * g[tid]       + bb[tid];
  float y1 = d1 * inv * g[tid + 256] + bb[tid + 256];
  size_t base = (size_t)t * 512;
  out32[base + tid] = y0;        out32[base + tid + 256] = y1;
  out16[base + tid] = (_Float16)y0; out16[base + tid + 256] = (_Float16)y1;
}

// ---------------- row L2-normalize -> f16 ----------------
__global__ void k_rownorm(const float* __restrict__ in, _Float16* __restrict__ out){
  __shared__ float red[256];
  int t = blockIdx.x, tid = threadIdx.x;
  const float* row = in + (size_t)t * 512;
  float x0 = row[tid], x1 = row[tid + 256];
  red[tid] = x0*x0 + x1*x1; __syncthreads();
  for (int o = 128; o > 0; o >>= 1){ if (tid < o) red[tid] += red[tid + o]; __syncthreads(); }
  float inv = 1.f / (sqrtf(red[0]) + 1e-13f);
  size_t base = (size_t)t * 512;
  out[base + tid] = (_Float16)(x0 * inv);
  out[base + tid + 256] = (_Float16)(x1 * inv);
}

// ---------------- generic batched WMMA GEMM with async double-buffered B tile ----------------
// C[b] = A[b](MxK f16, lda) x Wt(NxK f16, ldw)^T (+bias)(+resid fp32)(relu) -> C32/C16 (ldc)
// block = 256 thr = 8 waves stacked in M (128 rows) x 64 N cols.
// B tile (64n x 32k) staged in LDS (pad 40 halves/row for bank spread), double buffered;
// each thread issues exactly one async b128 per stage. A fragments: direct global b128,
// register-prefetched one K-step ahead.
__global__ void __launch_bounds__(256)
k_gemm(const _Float16* __restrict__ A, long long sA, int lda,
       const _Float16* __restrict__ Wt, long long sW, int ldw,
       const float* __restrict__ bias,
       const float* __restrict__ resid, long long sR,
       float* __restrict__ C32, _Float16* __restrict__ C16, long long sC, int ldc,
       int M, int N, int Kd, int relu){
  __shared__ _Float16 sB[2][64][40];
  int tid = threadIdx.x;
  int wave = tid >> 5, lane = tid & 31;
  int m0 = blockIdx.x * 128 + wave * 16;
  int n0 = blockIdx.y * 64;
  int b  = blockIdx.z;
  const _Float16* Ab = A  + (size_t)b * sA;
  const _Float16* Wb = Wt + (size_t)b * sW;
  int m0r = (m0 < M) ? m0 : 0;           // OOB waves read valid rows, discard results
  // stage mapping: 64 rows x 4 chunks of 8 halves
  int sn = tid >> 2, sc = (tid & 3) << 3;
  const _Float16* gsrc = Wb + (size_t)(n0 + sn) * ldw + sc;
  stage_b128(&sB[0][sn][sc], gsrc);      // k=0 tile
  stage_drain();
  __syncthreads();

  v8f acc[4];
#pragma unroll
  for (int j = 0; j < 4; ++j) acc[j] = zero8();
  v16h af = load_a(Ab + (size_t)m0r * lda, lda, lane);

  int buf = 0;
  for (int k0 = 0; k0 < Kd; k0 += 32){
    int kn = k0 + 32;
    bool more = kn < Kd;
    if (more) stage_b128(&sB[buf ^ 1][sn][sc], gsrc + kn);   // fill other buffer async
    v16h af_cur = af;
    if (more) af = load_a(Ab + (size_t)m0r * lda + kn, lda, lane);
#pragma unroll
    for (int j = 0; j < 4; ++j){
      v16h bf = load_b_nk(&sB[buf][j * 16][0], 40, lane);
      acc[j] = wmma16(af_cur, bf, acc[j]);
    }
    stage_drain();        // my async fills for buf^1 landed
    __syncthreads();      // all waves done reading buf; all fills visible
    buf ^= 1;
  }

  if (m0 < M){
    int nn = lane & 15, mb = (lane >> 4) << 3;
    const float* Rb = resid ? resid + (size_t)b * sR : nullptr;
    float*    Cb32 = C32 ? C32 + (size_t)b * sC : nullptr;
    _Float16* Cb16 = C16 ? C16 + (size_t)b * sC : nullptr;
#pragma unroll
    for (int j = 0; j < 4; ++j){
      int n = n0 + j*16 + nn;
      float bv = bias ? bias[n] : 0.f;
#pragma unroll
      for (int r = 0; r < 8; ++r){
        int m = m0 + mb + r;
        float v = acc[j][r] + bv;
        if (Rb) v += Rb[(size_t)m * ldc + n];
        if (relu) v = fmaxf(v, 0.f);
        size_t idx = (size_t)m * ldc + n;
        if (Cb32) Cb32[idx] = v;
        if (Cb16) Cb16[idx] = (_Float16)v;
      }
    }
  }
}

// ---------------- fused attention ----------------
// qkv: [B*S, 1536] f16 (Q|K|V). block = (qtile16, head, batch), 256 thr = 8 waves.
// scores in LDS (fp32) -> softmax -> P f16 in LDS -> P x V via WMMA -> ctx [B*S, 512] f16
#define AT_MAXS 512
__global__ void __launch_bounds__(256)
k_attn(const _Float16* __restrict__ qkv, const float* __restrict__ mask,
       _Float16* __restrict__ ctx, int S){
  __shared__ float    sS[16][AT_MAXS + 4];
  __shared__ _Float16 sP[16][AT_MAXS + 8];
  __shared__ float    sCtx[4][16][17];
  int qt = blockIdx.x, h = blockIdx.y, b = blockIdx.z;
  int wave = threadIdx.x >> 5, lane = threadIdx.x & 31;
  const int ld = 1536;
  const _Float16* Qb = qkv + (size_t)b * S * ld + h * 64;
  const _Float16* Kb = Qb + 512;
  const _Float16* Vb = Qb + 1024;
  int q0 = qt * 16;
  v16h qa0 = load_a(Qb + (size_t)q0 * ld,      ld, lane);   // hd 0..31
  v16h qa1 = load_a(Qb + (size_t)q0 * ld + 32, ld, lane);   // hd 32..63
  int nkt = S >> 4;
  for (int kt = wave; kt < nkt; kt += 8){
    v16h kb0 = load_b_nk(Kb + (size_t)(kt*16) * ld,      ld, lane);
    v16h kb1 = load_b_nk(Kb + (size_t)(kt*16) * ld + 32, ld, lane);
    v8f acc = zero8();
    acc = wmma16(qa0, kb0, acc);
    acc = wmma16(qa1, kb1, acc);
    int n = lane & 15, mb = (lane >> 4) << 3;
    int key = kt*16 + n;
    float mbias = (mask[(size_t)b * S + key] > 0.f) ? 0.f : -1e9f;
#pragma unroll
    for (int r = 0; r < 8; ++r) sS[mb + r][kt*16 + n] = acc[r] * 0.125f + mbias;
  }
  __syncthreads();
  { // softmax: 16 rows x 16 threads/row
    int row = threadIdx.x >> 4, sub = threadIdx.x & 15;
    float mx = -1e30f;
    for (int d = sub; d < S; d += 16) mx = fmaxf(mx, sS[row][d]);
    for (int o = 8; o >= 1; o >>= 1) mx = fmaxf(mx, __shfl_xor(mx, o, 16));
    float sm = 0.f;
    for (int d = sub; d < S; d += 16) sm += __expf(sS[row][d] - mx);
    for (int o = 8; o >= 1; o >>= 1) sm += __shfl_xor(sm, o, 16);
    float inv = 1.f / sm;
    for (int d = sub; d < S; d += 16) sP[row][d] = (_Float16)(__expf(sS[row][d] - mx) * inv);
  }
  __syncthreads();
  // P x V : wave -> (col tile c, key half)
  int c = wave & 3, khalf = wave >> 2;
  v8f acc = zero8();
  for (int j = khalf; j < (S >> 5); j += 2){
    int k0 = j * 32;
    v16h pa = load_a(&sP[0][0] + k0, AT_MAXS + 8, lane);
    v16h vb = load_b_kn(Vb + (size_t)k0 * ld + c*16, ld, lane);
    acc = wmma16(pa, vb, acc);
  }
  int n = lane & 15, mb = (lane >> 4) << 3;
  if (khalf == 1){
#pragma unroll
    for (int r = 0; r < 8; ++r) sCtx[c][mb + r][n] = acc[r];
  }
  __syncthreads();
  if (khalf == 0){
    _Float16* out = ctx + (size_t)(b * S + q0) * 512 + h * 64 + c * 16;
#pragma unroll
    for (int r = 0; r < 8; ++r){
      float v = acc[r] + sCtx[c][mb + r][n];
      out[(size_t)(mb + r) * 512 + n] = (_Float16)v;
    }
  }
}

// ---------------- RBF kernel pooling + position biases + dense ----------------
__global__ void __launch_bounds__(256)
k_pool(const float* __restrict__ cosm, const float* __restrict__ qmask,
       const float* __restrict__ dmask, const float* __restrict__ pos_bias,
       const float* __restrict__ pos_bias_abs, const float* __restrict__ dense_w,
       float* __restrict__ out){
  __shared__ float wgt[512];
  __shared__ float red[256];
  __shared__ float wsum[8][11];
  __shared__ float pk[11];
  int b = blockIdx.x, tid = threadIdx.x;
  float dl = 0.f;
  for (int d = tid; d < 512; d += 256) dl += dmask[(size_t)b*512 + d];
  red[tid] = dl; __syncthreads();
  for (int o = 128; o > 0; o >>= 1){ if (tid < o) red[tid] += red[tid + o]; __syncthreads(); }
  dl = red[0]; __syncthreads();
  int ri = (int)rintf(dl * 0.1f);
  for (int d = tid; d < 512; d += 256){
    float dm = dmask[(size_t)b*512 + d];
    int cs = (ri <= 0) ? 1 : ((d / ri + 1) > 10 ? 10 : (d / ri + 1));
    int pi = (int)((float)cs * dm);
    int ai = (int)((float)(d / 64 + 1) * dm);
    wgt[d] = pos_bias[pi] * pos_bias_abs[ai] * dm;
  }
  if (tid < 11) pk[tid] = 0.f;
  __syncthreads();
  int wv = tid >> 5, ln = tid & 31;
  for (int q = 0; q < 32; ++q){
    float acc[11];
#pragma unroll
    for (int k = 0; k < 11; ++k) acc[k] = 0.f;
    const float* crow = cosm + ((size_t)b*32 + q) * 512;
    for (int d = tid; d < 512; d += 256){
      float cv = crow[d], wd = wgt[d];
#pragma unroll
      for (int k = 0; k < 11; ++k){ float df = cv - c_MU[k]; acc[k] += __expf(-df*df*c_I2S[k]) * wd; }
    }
#pragma unroll
    for (int k = 0; k < 11; ++k)
      for (int o = 16; o >= 1; o >>= 1) acc[k] += __shfl_xor(acc[k], o, 32);
    if (ln == 0){
#pragma unroll
      for (int k = 0; k < 11; ++k) wsum[wv][k] = acc[k];
    }
    __syncthreads();
    if (tid < 11){
      float s = 0.f;
      for (int w8 = 0; w8 < 8; ++w8) s += wsum[w8][tid];
      pk[tid] += logf(fmaxf(s, 1e-10f)) * qmask[(size_t)b*32 + q];
    }
    __syncthreads();
  }
  if (tid == 0){
    float o = 0.f;
    for (int k = 0; k < 11; ++k) o += pk[k] * dense_w[k];
    out[b] = o;
  }
}

// ======================= host driver =======================
extern "C" void kernel_launch(void* const* d_in, const int* in_sizes, int n_in,
                              void* d_out, int out_size, void* d_ws, size_t ws_size,
                              hipStream_t stream){
  const float* qe  = (const float*)d_in[0];
  const float* de  = (const float*)d_in[1];
  const float* qm  = (const float*)d_in[2];
  const float* dm  = (const float*)d_in[3];
  const float* pos = (const float*)d_in[4];
  const float* lng = (const float*)d_in[5];
  const float* lnb = (const float*)d_in[6];
  const float* ipw = (const float*)d_in[7];
  const float* ipb = (const float*)d_in[8];
  const float* ow  = (const float*)d_in[9];
  const float* ob  = (const float*)d_in[10];
  const float* f1w = (const float*)d_in[11];
  const float* f1b = (const float*)d_in[12];
  const float* f2w = (const float*)d_in[13];
  const float* f2b = (const float*)d_in[14];
  const float* g1  = (const float*)d_in[15];
  const float* b1  = (const float*)d_in[16];
  const float* g2  = (const float*)d_in[17];
  const float* b2  = (const float*)d_in[18];
  const float* dw  = (const float*)d_in[19];
  const float* pb  = (const float*)d_in[20];
  const float* pba = (const float*)d_in[21];
  float* out = (float*)d_out;

  const int Bsz = 128, QL = 32, DL = 512, E = 512, FF = 2048, L3E = 1536;
  const size_t Td = (size_t)Bsz * DL;   // 65536 doc tokens
  const size_t Tq = (size_t)Bsz * QL;   // 4096 query tokens

  char* wsp = (char*)d_ws; size_t off = 0;
  auto alloc = [&](size_t bytes)->void*{ void* p = wsp + off; off += (bytes + 255) & ~(size_t)255; return p; };
  _Float16* wqkv = (_Float16*)alloc(2ull*L3E*E*2);
  _Float16* wout = (_Float16*)alloc(2ull*E*E*2);
  _Float16* wff1 = (_Float16*)alloc(2ull*FF*E*2);
  _Float16* wff2 = (_Float16*)alloc(2ull*E*FF*2);
  float*    x32  = (float*)   alloc(Td*E*4);
  _Float16* x16  = (_Float16*)alloc(Td*E*2);
  float*    y32  = (float*)   alloc(Td*E*4);
  _Float16* big  = (_Float16*)alloc(Td*FF*2);   // qkv / FF-hidden (aliased in time)
  _Float16* ctx  = (_Float16*)alloc(Td*E*2);
  _Float16* qn   = (_Float16*)alloc(Tq*E*2);
  _Float16* dn   = (_Float16*)y32;              // alias: y32 dead after final doc LN
  float*    cosb = (float*)big;                 // alias: big dead after encoders

  auto cvt = [&](const float* s, _Float16* d, long long n){
    int blocks = (int)((n + 2047) / 2048);
    k_f32_to_f16<<<blocks, 256, 0, stream>>>(s, d, n);
  };
  cvt(ipw, wqkv, 2ll*L3E*E);
  cvt(ow,  wout, 2ll*E*E);
  cvt(f1w, wff1, 2ll*FF*E);
  cvt(f2w, wff2, 2ll*E*FF);

  auto gemm = [&](const _Float16* A, long long sA, int lda,
                  const _Float16* Wt, long long sW, int ldw,
                  const float* bias, const float* resid, long long sR,
                  float* C32, _Float16* C16, long long sC, int ldc,
                  int M, int N, int K, int relu, int batch){
    dim3 g((M + 127) / 128, N / 64, batch);
    k_gemm<<<g, 256, 0, stream>>>(A, sA, lda, Wt, sW, ldw, bias, resid, sR,
                                  C32, C16, sC, ldc, M, N, K, relu);
  };

  auto encoder = [&](const float* emb, const float* msk, int S){
    int T = Bsz * S;
    k_addpos_ln<<<T, 256, 0, stream>>>(emb, pos, S, lng, lnb, x32, x16);
    for (int l = 0; l < 2; ++l){
      // QKV projection -> big [T,1536] f16
      gemm(x16, 0, E, wqkv + (size_t)l*L3E*E, 0, E, ipb + l*L3E,
           nullptr, 0, nullptr, big, 0, L3E, T, L3E, E, 0, 1);
      // attention -> ctx [T,512] f16
      k_attn<<<dim3(S/16, 8, Bsz), 256, 0, stream>>>(big, msk, ctx, S);
      // out proj + residual -> y32
      gemm(ctx, 0, E, wout + (size_t)l*E*E, 0, E, ob + l*E,
           x32, 0, y32, nullptr, 0, E, T, E, E, 0, 1);
      k_addpos_ln<<<T, 256, 0, stream>>>(y32, nullptr, S, g1 + l*E, b1 + l*E, x32, x16);
      // FF1 (relu) -> big [T,2048] f16
      gemm(x16, 0, E, wff1 + (size_t)l*FF*E, 0, E, f1b + l*FF,
           nullptr, 0, nullptr, big, 0, FF, T, FF, E, 1, 1);
      // FF2 + residual -> y32
      gemm(big, 0, FF, wff2 + (size_t)l*E*FF, 0, FF, f2b + l*E,
           x32, 0, y32, nullptr, 0, E, T, E, FF, 0, 1);
      k_addpos_ln<<<T, 256, 0, stream>>>(y32, nullptr, S, g2 + l*E, b2 + l*E, x32, x16);
    }
  };

  encoder(qe, qm, QL);
  k_rownorm<<<(int)Tq, 256, 0, stream>>>(x32, qn);
  encoder(de, dm, DL);
  k_rownorm<<<(int)Td, 256, 0, stream>>>(x32, dn);

  // batched cosine matrix: cos[b] = qn[b] (32x512) x dn[b]^T (512x512) -> [32,512] fp32
  gemm(qn, (long long)QL*E, E, dn, (long long)DL*E, E, nullptr, nullptr, 0,
       cosb, nullptr, (long long)QL*DL, DL, QL, DL, E, 0, Bsz);

  k_pool<<<Bsz, 256, 0, stream>>>(cosb, qm, dm, pb, pba, dw, out);

  (void)in_sizes; (void)n_in; (void)out_size; (void)ws_size;
}
